// D_MPNNet_62517543961329
// MI455X (gfx1250) — compile-verified, matrix operand order checked by reading, not run on hardware
//
#include <hip/hip_runtime.h>

typedef __attribute__((ext_vector_type(16))) __bf16 v16bf;
typedef __attribute__((ext_vector_type(8)))  __bf16 v8bf;
typedef __attribute__((ext_vector_type(8)))  float  v8f;

#define H 128
#define WS 136   // LDS row stride (ushorts) for W^T tile
#define AS 136   // LDS row stride (ushorts) for A tile
#define TPW 16   // tiles (of 16 rows) processed per wave

// float -> bf16 (round to nearest even)
__device__ __forceinline__ unsigned short f2bf(float f) {
    unsigned int u = __float_as_uint(f);
    u = (u + 0x7FFFu + ((u >> 16) & 1u)) >> 16;
    return (unsigned short)u;
}

// Wave-local LDS producer->consumer sync (per-wave private LDS region):
// forces s_wait_dscnt + a scheduling barrier without a workgroup barrier.
__device__ __forceinline__ void wave_lds_sync() {
    __builtin_amdgcn_fence(__ATOMIC_ACQ_REL, "workgroup");
    __builtin_amdgcn_wave_barrier();
}

// Load one 16-element bf16 fragment per lane from an LDS row.
// k0 = K base for this lane's low 8 elements; high 8 come from k0+16.
__device__ __forceinline__ v16bf ldfrag(const unsigned short* row, int k0) {
    v8bf lo = *(const v8bf*)(row + k0);
    v8bf hi = *(const v8bf*)(row + k0 + 16);
    v16bf r;
#pragma unroll
    for (int i = 0; i < 8; ++i) { r[i] = lo[i]; r[i + 8] = hi[i]; }
    return r;
}

// Cooperative (128-thread) load of W[128][128] transposed+converted into LDS,
// two bf16 packed per b32 DS write.
__device__ __forceinline__ void load_wT(unsigned short* lsW, const float* __restrict__ W) {
    int t = threadIdx.x;
#pragma unroll 4
    for (int k = 0; k < H; k += 2) {
        unsigned int p = (unsigned int)f2bf(W[k * H + t]) |
                         ((unsigned int)f2bf(W[(k + 1) * H + t]) << 16);
        *(unsigned int*)(lsW + t * WS + k) = p;
    }
}

// ---------------- x = h @ W_emb  ([N,28]@[28,128]) ----------------
__global__ void k_embed(const float* __restrict__ h, const float* __restrict__ Wemb,
                        float* __restrict__ x, int Nn) {
    int idx = blockIdx.x * blockDim.x + threadIdx.x;
    int node = idx >> 5;
    if (node >= Nn) return;
    int q = (idx & 31) * 4;
    float a0 = 0.f, a1 = 0.f, a2 = 0.f, a3 = 0.f;
    int hb = node * 28;
#pragma unroll 7
    for (int k = 0; k < 28; ++k) {
        float hv = h[hb + k];
        float4 w = ((const float4*)Wemb)[(k * H + q) >> 2];
        a0 += hv * w.x; a1 += hv * w.y; a2 += hv * w.z; a3 += hv * w.w;
    }
    float4 o = make_float4(a0, a1, a2, a3);
    ((float4*)x)[(node * H + q) >> 2] = o;
}

// ------- he = relu(cat(x[src], e) @ W_init + b_init)  (WMMA bf16) -------
__global__ __launch_bounds__(128) void k_init(const float* __restrict__ x,
                                              const float* __restrict__ e,
                                              const float* __restrict__ Winit,
                                              const float* __restrict__ binit,
                                              const int* __restrict__ src,
                                              float* __restrict__ he, int numTiles) {
    __shared__ unsigned short lsW[H * WS];      // W_init[0:128]^T in bf16
    __shared__ unsigned short lsA[4 * 16 * AS]; // per-wave A tiles (bf16)
    __shared__ float lsB[H];
    __shared__ float lsWE[6 * H];               // W_init[128:134] rows (f32)
    __shared__ float lsE[4 * 16 * 6];           // per-wave e tiles

    int t = threadIdx.x;
    load_wT(lsW, Winit);
    lsB[t] = binit[t];
#pragma unroll
    for (int j = 0; j < 6; ++j) lsWE[j * H + t] = Winit[(H + j) * H + t];
    __syncthreads();

    int wv = t >> 5, ln = t & 31;
    int r    = ln & 15;
    int ch   = (ln >> 4) * 64;
    int koff = (ln < 16) ? 0 : 8;
    int r0   = (ln < 16) ? 0 : 8;
    unsigned short* A = lsA + wv * 16 * AS;
    int tile0 = (blockIdx.x * 4 + wv) * TPW;

    for (int it = 0; it < TPW; ++it) {
        int tile = tile0 + it;            // wave-uniform
        if (tile >= numTiles) break;
        int eb = tile * 16;
        int eidx = eb + r;
        int abase = (src[eidx] * H + ch) >> 2;            // float4 index
#pragma unroll
        for (int q = 0; q < 16; ++q) {
            float4 a4 = ((const float4*)x)[abase + q];
            unsigned int lo = (unsigned int)f2bf(a4.x) | ((unsigned int)f2bf(a4.y) << 16);
            unsigned int hi = (unsigned int)f2bf(a4.z) | ((unsigned int)f2bf(a4.w) << 16);
            uint2 pk; pk.x = lo; pk.y = hi;
            *(uint2*)(A + r * AS + ch + q * 4) = pk;
        }
        if (ln < 16) {
#pragma unroll
            for (int k = 0; k < 6; ++k) lsE[(wv * 16 + r) * 6 + k] = e[eidx * 6 + k];
        }
        wave_lds_sync();

        v16bf af[4];
#pragma unroll
        for (int c = 0; c < 4; ++c) af[c] = ldfrag(A + r * AS, c * 32 + koff);

#pragma unroll
        for (int n = 0; n < 8; ++n) {
            v8f acc;
#pragma unroll
            for (int i = 0; i < 8; ++i) acc[i] = 0.f;
#pragma unroll
            for (int c = 0; c < 4; ++c) {
                v16bf bf = ldfrag(lsW + (n * 16 + r) * WS, c * 32 + koff);
                acc = __builtin_amdgcn_wmma_f32_16x16x32_bf16(false, af[c], false, bf,
                                                              (short)0, acc, false, false);
            }
            int col = n * 16 + r;
            float bb = lsB[col];
            int ob = (eb + r0) * H + col;                 // 32-bit element offset
#pragma unroll
            for (int rr = 0; rr < 8; ++rr) {
                int row = r0 + rr;
                float ea = 0.f;
#pragma unroll
                for (int k = 0; k < 6; ++k)
                    ea += lsE[(wv * 16 + row) * 6 + k] * lsWE[k * H + col];
                float v = acc[rr] + bb + ea;
                v = v > 0.f ? v : 0.f;
                he[ob + rr * H] = v;
            }
        }
        wave_lds_sync();   // A/E reused next iteration
    }
}

// ---------------- agg[dst[e]] += he[e]  (f32 atomics) ----------------
__global__ void k_scatter(const float* __restrict__ he, const int* __restrict__ dst,
                          float* __restrict__ agg, int Ee) {
    int idx = blockIdx.x * blockDim.x + threadIdx.x;
    int ed = idx >> 5;
    if (ed >= Ee) return;
    int q = (idx & 31) * 4;
    float4 v = ((const float4*)he)[(ed * H + q) >> 2];
    int p = dst[ed] * H + q;
    atomicAdd(&agg[p + 0], v.x); atomicAdd(&agg[p + 1], v.y);
    atomicAdd(&agg[p + 2], v.z); atomicAdd(&agg[p + 3], v.w);
}

// ----- he = relu((agg[src] - he[rev]) @ W + b) + he  (in-place, WMMA) -----
__global__ __launch_bounds__(128) void k_layer(const float* __restrict__ agg,
                                               float* __restrict__ he,
                                               const int* __restrict__ src,
                                               const float* __restrict__ W,
                                               const float* __restrict__ bias,
                                               int numTiles) {
    __shared__ unsigned short lsW[H * WS];
    __shared__ unsigned short lsA[4 * 16 * AS];
    __shared__ float lsB[H];

    int t = threadIdx.x;
    load_wT(lsW, W);
    lsB[t] = bias[t];
    __syncthreads();

    int wv = t >> 5, ln = t & 31;
    int r    = ln & 15;
    int ch   = (ln >> 4) * 64;
    int koff = (ln < 16) ? 0 : 8;
    int r0   = (ln < 16) ? 0 : 8;
    unsigned short* A = lsA + wv * 16 * AS;
    int tile0 = (blockIdx.x * 4 + wv) * TPW;

    for (int it = 0; it < TPW; ++it) {
        int tile = tile0 + it;            // wave-uniform
        if (tile >= numTiles) break;
        int eb = tile * 16;
        int eidx = eb + r;
        int ab = (src[eidx] * H + ch) >> 2;               // float4 index into agg
        int hb = ((eidx ^ 1) * H + ch) >> 2;              // float4 index into he

        // prefetch next tile's gathered rows behind this tile's work
        if (tile + 1 < numTiles) {
            int enx = eb + 16 + r;
            __builtin_prefetch(agg + src[enx] * H + ch, 0, 1);
            __builtin_prefetch(he + (enx ^ 1) * H + ch, 0, 1);
        }

#pragma unroll
        for (int q = 0; q < 16; ++q) {
            float4 a4 = ((const float4*)agg)[ab + q];
            float4 h4 = ((const float4*)he)[hb + q];
            unsigned int lo = (unsigned int)f2bf(a4.x - h4.x) |
                              ((unsigned int)f2bf(a4.y - h4.y) << 16);
            unsigned int hi = (unsigned int)f2bf(a4.z - h4.z) |
                              ((unsigned int)f2bf(a4.w - h4.w) << 16);
            uint2 pk; pk.x = lo; pk.y = hi;
            *(uint2*)(A + r * AS + ch + q * 4) = pk;
        }
        wave_lds_sync();

        v16bf af[4];
#pragma unroll
        for (int c = 0; c < 4; ++c) af[c] = ldfrag(A + r * AS, c * 32 + koff);

#pragma unroll
        for (int n = 0; n < 8; ++n) {
            v8f acc;
#pragma unroll
            for (int i = 0; i < 8; ++i) acc[i] = 0.f;
#pragma unroll
            for (int c = 0; c < 4; ++c) {
                v16bf bf = ldfrag(lsW + (n * 16 + r) * WS, c * 32 + koff);
                acc = __builtin_amdgcn_wmma_f32_16x16x32_bf16(false, af[c], false, bf,
                                                              (short)0, acc, false, false);
            }
            int col = n * 16 + r;
            float bb = lsB[col];
            int ob = (eb + r0) * H + col;                 // 32-bit element offset
#pragma unroll
            for (int rr = 0; rr < 8; ++rr) {
                float old = he[ob + rr * H];
                float v = acc[rr] + bb;
                v = v > 0.f ? v : 0.f;
                he[ob + rr * H] = v + old;
            }
        }
        wave_lds_sync();   // A reused next iteration
    }
}

// ---------------- hro = h_out @ W_ro  (dense rows, WMMA) ----------------
__global__ __launch_bounds__(128) void k_hro(const float* __restrict__ hin,
                                             const float* __restrict__ W,
                                             float* __restrict__ hout, int numTiles) {
    __shared__ unsigned short lsW[H * WS];
    __shared__ unsigned short lsA[4 * 16 * AS];

    int t = threadIdx.x;
    load_wT(lsW, W);
    __syncthreads();

    int wv = t >> 5, ln = t & 31;
    int r    = ln & 15;
    int ch   = (ln >> 4) * 64;
    int koff = (ln < 16) ? 0 : 8;
    int r0   = (ln < 16) ? 0 : 8;
    unsigned short* A = lsA + wv * 16 * AS;
    int tile0 = (blockIdx.x * 4 + wv) * TPW;

    for (int it = 0; it < TPW; ++it) {
        int tile = tile0 + it;            // wave-uniform
        if (tile >= numTiles) break;
        int nb = tile * 16;
        int ab = ((nb + r) * H + ch) >> 2;                // float4 index
#pragma unroll
        for (int q = 0; q < 16; ++q) {
            float4 a4 = ((const float4*)hin)[ab + q];
            unsigned int lo = (unsigned int)f2bf(a4.x) | ((unsigned int)f2bf(a4.y) << 16);
            unsigned int hi = (unsigned int)f2bf(a4.z) | ((unsigned int)f2bf(a4.w) << 16);
            uint2 pk; pk.x = lo; pk.y = hi;
            *(uint2*)(A + r * AS + ch + q * 4) = pk;
        }
        wave_lds_sync();

        v16bf af[4];
#pragma unroll
        for (int c = 0; c < 4; ++c) af[c] = ldfrag(A + r * AS, c * 32 + koff);

#pragma unroll
        for (int n = 0; n < 8; ++n) {
            v8f acc;
#pragma unroll
            for (int i = 0; i < 8; ++i) acc[i] = 0.f;
#pragma unroll
            for (int c = 0; c < 4; ++c) {
                v16bf bf = ldfrag(lsW + (n * 16 + r) * WS, c * 32 + koff);
                acc = __builtin_amdgcn_wmma_f32_16x16x32_bf16(false, af[c], false, bf,
                                                              (short)0, acc, false, false);
            }
            int col = n * 16 + r;
            int ob = (nb + r0) * H + col;
#pragma unroll
            for (int rr = 0; rr < 8; ++rr)
                hout[ob + rr * H] = acc[rr];
        }
        wave_lds_sync();   // A reused next iteration
    }
}

// ---------------- hg[gid[n]] += hro[n]  (f32 atomics) ----------------
__global__ void k_graphsum(const float* __restrict__ hro, const int* __restrict__ gid,
                           float* __restrict__ hg, int Nn) {
    int idx = blockIdx.x * blockDim.x + threadIdx.x;
    int node = idx >> 5;
    if (node >= Nn) return;
    int q = (idx & 31) * 4;
    float4 v = ((const float4*)hro)[(node * H + q) >> 2];
    int p = gid[node] * H + q;
    atomicAdd(&hg[p + 0], v.x); atomicAdd(&hg[p + 1], v.y);
    atomicAdd(&hg[p + 2], v.z); atomicAdd(&hg[p + 3], v.w);
}

// ---------------- out = hg @ W_pred + b_pred ----------------
__global__ void k_pred(const float* __restrict__ hg, const float* __restrict__ Wp,
                       const float* __restrict__ bp, float* __restrict__ out) {
    __shared__ float red[128];
    int b = blockIdx.x, t = threadIdx.x;
    red[t] = hg[b * H + t] * Wp[t];
    __syncthreads();
    for (int s = 64; s > 0; s >>= 1) {
        if (t < s) red[t] += red[t + s];
        __syncthreads();
    }
    if (t == 0) out[b] = red[0] + bp[0];
}

extern "C" void kernel_launch(void* const* d_in, const int* in_sizes, int n_in,
                              void* d_out, int out_size, void* d_ws, size_t ws_size,
                              hipStream_t stream) {
    const float* h     = (const float*)d_in[0];
    const float* e     = (const float*)d_in[1];
    const float* Wemb  = (const float*)d_in[2];
    const float* Winit = (const float*)d_in[3];
    const float* binit = (const float*)d_in[4];
    const float* Wlay  = (const float*)d_in[5];
    const float* blay  = (const float*)d_in[6];
    const float* Wro   = (const float*)d_in[7];
    const float* Wpred = (const float*)d_in[8];
    const float* bpred = (const float*)d_in[9];
    const int*   src   = (const int*)d_in[10];
    const int*   dst   = (const int*)d_in[11];
    const int*   gid   = (const int*)d_in[12];

    const int N = 200000, E = 800000, B = 2000, L = 4;
    const int eTiles = E / 16;                    // 50000
    const int nTiles = N / 16;                    // 12500
    const int tilesPerWG = 4 * TPW;               // 64
    const int eWGs = (eTiles + tilesPerWG - 1) / tilesPerWG;
    const int nWGs = (nTiles + tilesPerWG - 1) / tilesPerWG;

    float* x   = (float*)d_ws;                 // N*H
    float* he  = x   + (size_t)N * H;          // E*H
    float* agg = he  + (size_t)E * H;          // N*H
    float* hg  = agg + (size_t)N * H;          // B*H
    float* hro = x;                            // reuse x after k_init

    // 1. node embedding
    k_embed<<<(N * 32 + 255) / 256, 256, 0, stream>>>(h, Wemb, x, N);
    // 2. edge-state init
    k_init<<<eWGs, 128, 0, stream>>>(x, e, Winit, binit, src, he, eTiles);
    // 3. D-MPNN layers
    for (int l = 0; l < L; ++l) {
        hipMemsetAsync(agg, 0, (size_t)N * H * sizeof(float), stream);
        k_scatter<<<(E * 32 + 255) / 256, 256, 0, stream>>>(he, dst, agg, E);
        k_layer<<<eWGs, 128, 0, stream>>>(agg, he, src,
                                          Wlay + (size_t)l * H * H, blay + (size_t)l * H,
                                          eTiles);
    }
    // 4. final node aggregation
    hipMemsetAsync(agg, 0, (size_t)N * H * sizeof(float), stream);
    k_scatter<<<(E * 32 + 255) / 256, 256, 0, stream>>>(he, dst, agg, E);
    // 5. readout linear
    k_hro<<<nWGs, 128, 0, stream>>>(agg, Wro, hro, nTiles);
    // 6. per-graph sum
    hipMemsetAsync(hg, 0, (size_t)B * H * sizeof(float), stream);
    k_graphsum<<<(N * 32 + 255) / 256, 256, 0, stream>>>(hro, gid, hg, N);
    // 7. prediction head
    k_pred<<<B, 128, 0, stream>>>(hg, Wpred, bpred, (float*)d_out);
}
